// MultiHeadAttention_75213467287984
// MI455X (gfx1250) — matrix operand order repaired and checked by reading, not compiled, not run on hardware
//
#include <hip/hip_runtime.h>
#include <math.h>

// Shapes (fixed by the reference)
#define BB 4
#define SS 2048
#define DD 1024
#define HH 16
#define HD 64
// M = BB*SS = 8192 rows for all projections

typedef __attribute__((ext_vector_type(16))) __bf16 v16bf;
typedef __attribute__((ext_vector_type(8)))  float  v8f;
typedef __attribute__((ext_vector_type(4)))  unsigned int u32x4;
typedef __attribute__((ext_vector_type(8)))  int          i32x8;
typedef __attribute__((ext_vector_type(4)))  int          i32x4;

struct __attribute__((aligned(16))) F4 { float x, y, z, w; };
struct __attribute__((aligned(16))) U4 { unsigned int x, y, z, w; };
struct __attribute__((aligned(8)))  U2 { unsigned int x, y; };

union Frag {
    v16bf        v;
    unsigned int u[8];
};

// TDM available? (device pass only; 5-arg on clang-22/ROCm7.2, 6-arg on clang-23)
#if defined(__gfx1250__) && __has_builtin(__builtin_amdgcn_tensor_load_to_lds) && \
    __has_builtin(__builtin_amdgcn_s_wait_tensorcnt)
#define USE_TDM 1
#else
#define USE_TDM 0
#endif

__device__ __forceinline__ unsigned short f2bf(float f) {
    union { float f; unsigned int u; } c; c.f = f;
    unsigned int u = c.u;
    u += 0x7FFFu + ((u >> 16) & 1u);   // round-to-nearest-even
    return (unsigned short)(u >> 16);
}
__device__ __forceinline__ unsigned int pack2bf(float lo, float hi) {
    return (unsigned int)f2bf(lo) | ((unsigned int)f2bf(hi) << 16);
}

// ---------------------------------------------------------------------------
// GEMM:  C[m,n] = sum_k A[m,k] * W[n,k] + bias[n]      (X @ W^T + b)
// MODE 0: A fp32 [M,DD], out bf16 scattered to [B,H,S,HD]
// MODE 1: A bf16 [M,DD], out fp32 [M,DD] linear
// Block: 128 thr = 4 waves; block tile 64x64; wave tile 32x32 (2x2 WMMAs);
// K-step 32, double-buffered LDS (1 barrier/iter, global latency hidden).
// ---------------------------------------------------------------------------
template <int MODE>
__global__ __launch_bounds__(128)
void gemm_bias_kernel(const void* __restrict__ Ap, const float* __restrict__ W,
                      const float* __restrict__ bias, void* __restrict__ outp) {
    __shared__ unsigned short ldsA[2][64 * 40];  // bf16, row stride 40 (8B aligned)
    __shared__ unsigned short ldsB[2][64 * 40];

    const int tid  = threadIdx.x;
    const int w    = tid >> 5;
    const int L    = tid & 31;
    const int half = L >> 4;
    const int lm   = L & 15;
    const int n0   = blockIdx.x * 64;
    const int m0   = blockIdx.y * 64;

    v8f acc[2][2] = {};

    // Prefetch registers for the next tile
    F4 pw[4];
    F4 pa32[4];
    U2 pa16[4];

    auto gload = [&](int k0) {
#pragma unroll
        for (int j = 0; j < 4; ++j) {
            int idx = tid * 4 + j;                   // 0..511 16B chunks
            int row = idx >> 3;
            int c4  = (idx & 7) * 4;
            if (MODE == 0)
                pa32[j] = *(const F4*)((const float*)Ap + (size_t)(m0 + row) * DD + k0 + c4);
            else
                pa16[j] = *(const U2*)((const unsigned short*)Ap + (size_t)(m0 + row) * DD + k0 + c4);
            pw[j] = *(const F4*)(W + (size_t)(n0 + row) * DD + k0 + c4);
        }
    };
    auto lstore = [&](int buf) {
#pragma unroll
        for (int j = 0; j < 4; ++j) {
            int idx = tid * 4 + j;
            int row = idx >> 3;
            int c4  = (idx & 7) * 4;
            if (MODE == 0) {
                U2 p; p.x = pack2bf(pa32[j].x, pa32[j].y); p.y = pack2bf(pa32[j].z, pa32[j].w);
                *(U2*)&ldsA[buf][row * 40 + c4] = p;
            } else {
                *(U2*)&ldsA[buf][row * 40 + c4] = pa16[j];
            }
            U2 q; q.x = pack2bf(pw[j].x, pw[j].y); q.y = pack2bf(pw[j].z, pw[j].w);
            *(U2*)&ldsB[buf][row * 40 + c4] = q;
        }
    };

    const int kIters = DD / 32;
    gload(0);
    for (int it = 0; it < kIters; ++it) {
        const int buf = it & 1;
        lstore(buf);
        __syncthreads();
        if (it + 1 < kIters) gload((it + 1) * 32);   // overlap with compute below

        Frag a[2], b[2];
#pragma unroll
        for (int sm = 0; sm < 2; ++sm) {
            int ml = (w >> 1) * 32 + sm * 16 + lm;   // A row (m = L%16 pattern)
#pragma unroll
            for (int d = 0; d < 8; ++d) {
                int k = ((d >> 2) << 4) + half * 8 + ((d & 3) << 1);
                a[sm].u[d] = *(const unsigned int*)&ldsA[buf][ml * 40 + k];
            }
        }
#pragma unroll
        for (int sn = 0; sn < 2; ++sn) {
            int nl = (w & 1) * 32 + sn * 16 + lm;    // B col (n = L%16 pattern)
#pragma unroll
            for (int d = 0; d < 8; ++d) {
                int k = ((d >> 2) << 4) + half * 8 + ((d & 3) << 1);
                b[sn].u[d] = *(const unsigned int*)&ldsB[buf][nl * 40 + k];
            }
        }
#pragma unroll
        for (int sm = 0; sm < 2; ++sm)
#pragma unroll
            for (int sn = 0; sn < 2; ++sn)
                acc[sm][sn] = __builtin_amdgcn_wmma_f32_16x16x32_bf16(
                    false, a[sm].v, false, b[sn].v, (short)0, acc[sm][sn], false, false);
    }

    // Epilogue: C layout lane L holds m = (L/16)*8 + v, n = L%16
#pragma unroll
    for (int sn = 0; sn < 2; ++sn) {
        int n = n0 + (w & 1) * 32 + sn * 16 + lm;
        float bv = bias[n];
#pragma unroll
        for (int sm = 0; sm < 2; ++sm) {
#pragma unroll
            for (int v = 0; v < 8; ++v) {
                int m = m0 + (w >> 1) * 32 + sm * 16 + half * 8 + v;
                float val = acc[sm][sn][v] + bv;
                if (MODE == 0) {
                    int b_ = m >> 11, s_ = m & (SS - 1);   // m = b*S + s
                    int h_ = n >> 6,  hd_ = n & 63;        // n = h*HD + hd
                    ((unsigned short*)outp)[(((size_t)(b_ * HH + h_) * SS + s_) << 6) + hd_] =
                        f2bf(val);
                } else {
                    ((float*)outp)[(size_t)m * DD + n] = val;
                }
            }
        }
    }
}

// ---------------------------------------------------------------------------
// Flash attention (causal): one block per (bh, 64-row q tile), 4 waves.
// Wave w owns q rows [q0 + w*16, +16). Q A-frags persist in registers.
// K staged LDS[n=kj][k=d] (TDM when available); V staged transposed [hd][kj].
// Softmax in base-2 domain (scale 0.125*log2e folded in, native v_exp_f32).
// ---------------------------------------------------------------------------
__global__ __launch_bounds__(128)
void flash_attn_kernel(const unsigned short* __restrict__ Qb,
                       const unsigned short* __restrict__ Kb,
                       const unsigned short* __restrict__ Vb,
                       unsigned short* __restrict__ Cb) {
    __shared__ unsigned short ldsK [64 * 80];  // K  [kj][d],  row stride 80 (160B)
    __shared__ unsigned short ldsVt[64 * 80];  // V^T [hd][kj]
    __shared__ unsigned short ldsP [64 * 80];  // P  [q][kj]

    const int tid  = threadIdx.x;
    const int w    = tid >> 5;
    const int L    = tid & 31;
    const int half = L >> 4;
    const int lm   = L & 15;
    const int q0   = blockIdx.x * 64;
    const int bh   = blockIdx.y;

    const float SCL = 0.125f * 1.4426950408889634f;  // 1/sqrt(HD) * log2(e)

    // Persistent Q A-fragments (HD=64 -> two 16x32 frags)
    Frag qa[2];
    {
        size_t base = ((size_t)bh * SS + q0 + w * 16 + lm) * HD;
#pragma unroll
        for (int dh = 0; dh < 2; ++dh) {
            U4 t0 = *(const U4*)(Qb + base + dh * 32 + half * 8);
            U4 t1 = *(const U4*)(Qb + base + dh * 32 + 16 + half * 8);
            qa[dh].u[0] = t0.x; qa[dh].u[1] = t0.y; qa[dh].u[2] = t0.z; qa[dh].u[3] = t0.w;
            qa[dh].u[4] = t1.x; qa[dh].u[5] = t1.y; qa[dh].u[6] = t1.z; qa[dh].u[7] = t1.w;
        }
    }

    float mrun[8], lrun[8];
    v8f   oacc[4] = {};
#pragma unroll
    for (int v = 0; v < 8; ++v) { mrun[v] = -__builtin_inff(); lrun[v] = 0.f; }

    const int nkb = (q0 >> 6) + 1;             // causal: only key blocks <= q block
    for (int kb = 0; kb < nkb; ++kb) {
        const int k0 = kb * 64;
        __syncthreads();
        // ---- Stage K tile (TDM if available) and transposed V tile ----
        {
            int r = tid >> 1, coff = (tid & 1) * 32;
            size_t gk = ((size_t)bh * SS + k0 + r) * HD;
#if USE_TDM
            if (tid < 32) {
                // 2D D#: 64x64 bf16 tile, row stride 64 elem; LDS pad 32B per 128B row
                unsigned long long ga =
                    (unsigned long long)(const void*)(Kb + ((size_t)bh * SS + k0) * HD);
                unsigned ldsoff = (unsigned)(size_t)(void*)&ldsK[0];
                u32x4 g0;
                g0.x = 1u;                                   // count=1 (user mode)
                g0.y = ldsoff;                               // lds_addr (bytes)
                g0.z = (unsigned)ga;                         // global_addr[31:0]
                g0.w = (unsigned)((ga >> 32) & 0x01FFFFFFu)  // global_addr[56:32]
                       | (2u << 30);                         // type = 2 (image)
                i32x8 g1 = {};
                g1[0] = (1 << 16)      // data_size = 2 bytes
                      | (1 << 20)      // pad_enable
                      | (4 << 22)      // pad_interval: 32 DWORDs (128B = one row)
                      | (7 << 25);     // pad_amount: 8 DWORDs (32B -> 160B row pitch)
                g1[1] = (int)(64u << 16);     // tensor_dim0 = 64 (low16 at [63:48])
                g1[2] = (int)(2048u << 16);   // tensor_dim1 = 2048 (low16 at [95:80])
                g1[3] = (int)(64u << 16);     // tile_dim0 = 64 ([127:112])
                g1[4] = 64;                   // tile_dim1 = 64 ([143:128])
                g1[5] = 64;                   // tensor_dim0_stride = 64 ([191:160])
                i32x4 z4 = {};
#if __clang_major__ >= 23
                i32x8 z8 = {};
                __builtin_amdgcn_tensor_load_to_lds(g0, g1, z4, z4, z8, 0);
#else
                __builtin_amdgcn_tensor_load_to_lds(g0, g1, z4, z4, 0);
#endif
            }
#endif
#pragma unroll
            for (int j = 0; j < 4; ++j) {
                int c = coff + j * 8;
#if !USE_TDM
                U4 kv = *(const U4*)(Kb + gk + c);
                *(U4*)&ldsK[r * 80 + c] = kv;
#endif
                U4 vv = *(const U4*)(Vb + gk + c);
                const unsigned short* vs = (const unsigned short*)&vv;
#pragma unroll
                for (int i = 0; i < 8; ++i) ldsVt[(c + i) * 80 + r] = vs[i];
            }
            if (kb + 1 < nkb) {                         // warm L2 for next tile
                __builtin_prefetch(Kb + gk + (size_t)64 * HD, 0, 0);
                __builtin_prefetch(Vb + gk + (size_t)64 * HD, 0, 0);
            }
#if USE_TDM
            if (tid < 32) __builtin_amdgcn_s_wait_tensorcnt(0);
#endif
        }
        __syncthreads();

        // ---- scores: S = Q K^T  (4 kj subtiles x 2 d-halves) ----
        v8f sacc[4] = {};
#pragma unroll
        for (int sn = 0; sn < 4; ++sn) {
#pragma unroll
            for (int dh = 0; dh < 2; ++dh) {
                Frag bfr;
                int nl = sn * 16 + lm;
#pragma unroll
                for (int d = 0; d < 8; ++d) {
                    int k = dh * 32 + ((d >> 2) << 4) + half * 8 + ((d & 3) << 1);
                    bfr.u[d] = *(const unsigned int*)&ldsK[nl * 80 + k];
                }
                sacc[sn] = __builtin_amdgcn_wmma_f32_16x16x32_bf16(
                    false, qa[dh].v, false, bfr.v, (short)0, sacc[sn], false, false);
            }
        }
        // ---- scale (base-2 domain) + causal mask ----
#pragma unroll
        for (int sn = 0; sn < 4; ++sn) {
            int kg = k0 + sn * 16 + lm;
#pragma unroll
            for (int v = 0; v < 8; ++v) {
                int qg = q0 + w * 16 + half * 8 + v;
                float s = sacc[sn][v] * SCL;
                sacc[sn][v] = (kg > qg) ? -__builtin_inff() : s;
            }
        }
        // ---- online softmax (exp2; row reductions across owning 16-lane half) ----
#pragma unroll
        for (int v = 0; v < 8; ++v) {
            float mx = fmaxf(fmaxf(sacc[0][v], sacc[1][v]), fmaxf(sacc[2][v], sacc[3][v]));
#pragma unroll
            for (int off = 1; off < 16; off <<= 1) mx = fmaxf(mx, __shfl_xor(mx, off, 32));
            float mn = fmaxf(mrun[v], mx);
            float f  = __builtin_amdgcn_exp2f(mrun[v] - mn);   // 0 on first block
            float rs = 0.f;
#pragma unroll
            for (int sn = 0; sn < 4; ++sn) {
                float p = __builtin_amdgcn_exp2f(sacc[sn][v] - mn);  // -inf -> 0
                sacc[sn][v] = p;
                rs += p;
            }
#pragma unroll
            for (int off = 1; off < 16; off <<= 1) rs += __shfl_xor(rs, off, 32);
            lrun[v] = lrun[v] * f + rs;
            mrun[v] = mn;
#pragma unroll
            for (int st = 0; st < 4; ++st) oacc[st][v] *= f;
        }
        // ---- P -> LDS (C layout -> memory) for A-frag reload ----
#pragma unroll
        for (int sn = 0; sn < 4; ++sn)
#pragma unroll
            for (int v = 0; v < 8; ++v)
                ldsP[(w * 16 + half * 8 + v) * 80 + sn * 16 + lm] = f2bf(sacc[sn][v]);
        __syncthreads();

        // ---- O += P @ V  (2 kj-halves x 4 hd subtiles) ----
#pragma unroll
        for (int kh = 0; kh < 2; ++kh) {
            Frag pa;
            int pr = w * 16 + lm;
#pragma unroll
            for (int d = 0; d < 8; ++d) {
                int kj = kh * 32 + ((d >> 2) << 4) + half * 8 + ((d & 3) << 1);
                pa.u[d] = *(const unsigned int*)&ldsP[pr * 80 + kj];
            }
#pragma unroll
            for (int st = 0; st < 4; ++st) {
                Frag vb;
                int nl = st * 16 + lm;
#pragma unroll
                for (int d = 0; d < 8; ++d) {
                    int kj = kh * 32 + ((d >> 2) << 4) + half * 8 + ((d & 3) << 1);
                    vb.u[d] = *(const unsigned int*)&ldsVt[nl * 80 + kj];
                }
                oacc[st] = __builtin_amdgcn_wmma_f32_16x16x32_bf16(
                    false, pa.v, false, vb.v, (short)0, oacc[st], false, false);
            }
        }
    }

    // Normalize and write concat [B*S, D] bf16 for the output projection
    const int b_ = bh >> 4, h_ = bh & 15;
#pragma unroll
    for (int v = 0; v < 8; ++v) {
        float inv = 1.0f / lrun[v];
        int srow  = q0 + w * 16 + half * 8 + v;
        size_t orow = (size_t)(b_ * SS + srow) * DD + h_ * HD;
#pragma unroll
        for (int st = 0; st < 4; ++st)
            Cb[orow + st * 16 + lm] = f2bf(oacc[st][v] * inv);
    }
}

// ---------------------------------------------------------------------------
extern "C" void kernel_launch(void* const* d_in, const int* in_sizes, int n_in,
                              void* d_out, int out_size, void* d_ws, size_t ws_size,
                              hipStream_t stream) {
    const float* query = (const float*)d_in[0];
    const float* key_  = (const float*)d_in[1];
    const float* value = (const float*)d_in[2];
    const float* Wq    = (const float*)d_in[3];
    const float* bq    = (const float*)d_in[4];
    const float* Wk    = (const float*)d_in[5];
    const float* bk    = (const float*)d_in[6];
    const float* Wv    = (const float*)d_in[7];
    const float* bv    = (const float*)d_in[8];
    const float* Wo    = (const float*)d_in[9];
    const float* bo    = (const float*)d_in[10];
    // d_in[11] is the tril mask; causality is applied analytically.

    const size_t elems = (size_t)BB * HH * SS * HD;  // 8,388,608 bf16 per buffer
    unsigned short* Qb = (unsigned short*)d_ws;
    unsigned short* Kb = Qb + elems;
    unsigned short* Vb = Kb + elems;
    unsigned short* Cb = Vb + elems;                 // total 64 MB of d_ws

    dim3 gg(DD / 64, (BB * SS) / 64);                // (16, 128)
    gemm_bias_kernel<0><<<gg, 128, 0, stream>>>(query, Wq, bq, Qb);
    gemm_bias_kernel<0><<<gg, 128, 0, stream>>>(key_,  Wk, bk, Kb);
    gemm_bias_kernel<0><<<gg, 128, 0, stream>>>(value, Wv, bv, Vb);

    flash_attn_kernel<<<dim3(SS / 64, BB * HH), 128, 0, stream>>>(Qb, Kb, Vb, Cb);

    gemm_bias_kernel<1><<<gg, 128, 0, stream>>>(Cb, Wo, bo, (float*)d_out);
}
